// RNN_73521250173348
// MI455X (gfx1250) — matrix operand order repaired and checked by reading, not compiled, not run on hardware
//
#include <hip/hip_runtime.h>

typedef __attribute__((ext_vector_type(16))) _Float16 v16h;
typedef __attribute__((ext_vector_type(8)))  float    v8f;

#define N_TRIALS 256
#define T_STEPS  1000
#define IN_DIM   64
#define SIZE     256
#define OUT_DIM  32
#define DT       0.2f

#define TB       16      // trials per block (recurrence kernel)
#define NTHREADS 512     // 16 wave32s

// ---------------- recurrence kernel LDS layout (halfs) ----------------
#define JOP_OFF   0                    // 256x256 f16, B-operand order (128 KB)
#define BOP_OFF   65536                // 256x64  f16, B-operand order (32 KB)
#define SOP_OFF   (65536 + 16384)      // 2 x (16x256) sigma(x), A-operand order (16 KB)
#define IOP_OFF   (81920 + 8192)       // 2 x (16x64) input tile, A-operand order (4 KB)
#define LDS1_HALFS (90112 + 2048)      // 92160 halfs = 184320 bytes

// Inverse map for CDNA5 16-bit A-matrix 16x32 operand layout:
// element K of row m -> half-index ((kk*32 + hi*16 + m)*16 + i), kk=K/32,
// K%32 in [0,8)->(hi=0,i=K%8), [8,16)->(hi=1,i=K%8),
// [16,24)->(hi=0,i=8+K%8), [24,32)->(hi=1,i=8+K%8)
__device__ __forceinline__ int a_addr(int m, int K) {
    int q  = (K & 31) >> 3;
    int hi = q & 1;
    int i  = (K & 7) + ((q >> 1) << 3);
    int kk = K >> 5;
    return ((kk * 32 + hi * 16 + m) << 4) + i;
}

__device__ __forceinline__ float sigmoidf(float x) {
    return 1.0f / (1.0f + __expf(-x));
}

// =====================================================================
// Kernel 1: the sequential recurrence. 16 blocks x 512 threads; each
// block owns 16 trials for all 1000 steps. ONE barrier per step.
// Writes x_seq only; readout y is a separate parallel pass.
// =====================================================================
__global__ __launch_bounds__(NTHREADS)
void rnn_recur_kernel(const float* __restrict__ inp,  // (256,1000,64)
                      const float* __restrict__ Bg,   // (256,64)
                      const float* __restrict__ Jg,   // (256,256)
                      float* __restrict__ out_x)      // (256,1001,256)
{
    extern __shared__ _Float16 smem[];
    _Float16* Jop = smem + JOP_OFF;
    _Float16* Bop = smem + BOP_OFF;
    _Float16* Sop = smem + SOP_OFF;   // two 4096-half buffers
    _Float16* Iop = smem + IOP_OFF;   // two 1024-half buffers

    const int tid  = threadIdx.x;
    const int lane = tid & 31;
    const int w    = tid >> 5;        // wave id 0..15 -> size-columns 16w..16w+15
    const int b0   = blockIdx.x * TB;

    // ---- one-time staging: weights -> LDS f16 in WMMA B-operand order ----
    for (int idx = tid; idx < SIZE * SIZE; idx += NTHREADS) {
        int n = idx >> 8, k = idx & 255;
        Jop[((n >> 4) << 12) + a_addr(n & 15, k)] = (_Float16)Jg[idx];
    }
    for (int idx = tid; idx < SIZE * IN_DIM; idx += NTHREADS) {
        int n = idx >> 6, k = idx & 63;
        Bop[((n >> 4) << 10) + a_addr(n & 15, k)] = (_Float16)Bg[idx];
    }
    // step-0 A operand: sigmoid(0) = 0.5
    for (int idx = tid; idx < 16 * SIZE; idx += NTHREADS) {
        Sop[idx] = (_Float16)0.5f;
    }
    // x_seq[:,0,:] = 0
    for (int idx = tid; idx < TB * SIZE; idx += NTHREADS) {
        int m = idx >> 8, n = idx & 255;
        out_x[(size_t)(b0 + m) * (T_STEPS + 1) * SIZE + n] = 0.0f;
    }

    const int im = w;                 // trial row this thread stages inputs for
    const float* irow = inp + (size_t)(b0 + im) * T_STEPS * IN_DIM;
    const int K0   = 2 * lane;
    const int ia0  = a_addr(im, K0);
    const int ia1  = a_addr(im, K0 + 1);

    // prologue: stage input tile for t = 0 into Iop[0]
    {
        float2 vv = *(const float2*)(irow + 2 * lane);
        Iop[ia0] = (_Float16)vv.x;
        Iop[ia1] = (_Float16)vv.y;
    }

    float xr[8] = {0.f, 0.f, 0.f, 0.f, 0.f, 0.f, 0.f, 0.f};
    const int mlo  = (lane < 16) ? 0 : 8;
    const int ncol = w * 16 + (lane & 15);

    for (int t = 0; t < T_STEPS; ++t) {
        __syncthreads();   // orders all iter-(t-1) LDS writes vs iter-t reads

        const _Float16* Icur = Iop + ((t & 1) << 10);
        _Float16*       Inxt = Iop + (((t + 1) & 1) << 10);
        const _Float16* Scur = Sop + ((t & 1) << 12);
        _Float16*       Snxt = Sop + (((t + 1) & 1) << 12);

        // issue next-step input load early (independent of the WMMA chain)
        const int tn = (t + 1 < T_STEPS) ? (t + 1) : t;   // clamp; avoids OOB
        float2 vv = *(const float2*)(irow + (size_t)tn * IN_DIM + 2 * lane);
        __builtin_prefetch(irow + (size_t)(tn + 1) * IN_DIM + 2 * lane, 0, 1);

        // ---- acc = inp_t @ B^T + sigma(x_t) @ J^T : 4 independent chains ----
        v8f c0 = {}, c1 = {}, c2 = {}, c3 = {};
        {
            v16h a = *(const v16h*)(Icur + lane * 16);
            v16h b = *(const v16h*)(Bop + w * 1024 + lane * 16);
            c0 = __builtin_amdgcn_wmma_f32_16x16x32_f16(false, a, false, b,
                                                        (short)0, c0, false, false);
        }
        {
            v16h a = *(const v16h*)(Icur + 512 + lane * 16);
            v16h b = *(const v16h*)(Bop + w * 1024 + 512 + lane * 16);
            c1 = __builtin_amdgcn_wmma_f32_16x16x32_f16(false, a, false, b,
                                                        (short)0, c1, false, false);
        }
#pragma unroll
        for (int kk = 0; kk < 8; kk += 4) {
            v16h a0 = *(const v16h*)(Scur + (kk + 0) * 512 + lane * 16);
            v16h b0 = *(const v16h*)(Jop + w * 4096 + (kk + 0) * 512 + lane * 16);
            c0 = __builtin_amdgcn_wmma_f32_16x16x32_f16(false, a0, false, b0,
                                                        (short)0, c0, false, false);
            v16h a1 = *(const v16h*)(Scur + (kk + 1) * 512 + lane * 16);
            v16h b1 = *(const v16h*)(Jop + w * 4096 + (kk + 1) * 512 + lane * 16);
            c1 = __builtin_amdgcn_wmma_f32_16x16x32_f16(false, a1, false, b1,
                                                        (short)0, c1, false, false);
            v16h a2 = *(const v16h*)(Scur + (kk + 2) * 512 + lane * 16);
            v16h b2 = *(const v16h*)(Jop + w * 4096 + (kk + 2) * 512 + lane * 16);
            c2 = __builtin_amdgcn_wmma_f32_16x16x32_f16(false, a2, false, b2,
                                                        (short)0, c2, false, false);
            v16h a3 = *(const v16h*)(Scur + (kk + 3) * 512 + lane * 16);
            v16h b3 = *(const v16h*)(Jop + w * 4096 + (kk + 3) * 512 + lane * 16);
            c3 = __builtin_amdgcn_wmma_f32_16x16x32_f16(false, a3, false, b3,
                                                        (short)0, c3, false, false);
        }
        v8f c = (c0 + c1) + (c2 + c3);

        // stage next-step input tile (buffer read only after next barrier)
        Inxt[ia0] = (_Float16)vv.x;
        Inxt[ia1] = (_Float16)vv.y;

        // ---- state update, write x_seq, stage sigma(x_{t+1}) ----
#pragma unroll
        for (int v = 0; v < 8; ++v) {
            float xn = (1.0f - DT) * xr[v] + DT * c[v];
            xr[v] = xn;
            int m = v + mlo;
            out_x[(size_t)(b0 + m) * (T_STEPS + 1) * SIZE
                  + (size_t)(t + 1) * SIZE + ncol] = xn;
            Snxt[a_addr(m, ncol)] = (_Float16)sigmoidf(xn);
        }
    }
}

// =====================================================================
// Kernel 2: fully parallel readout  y = sigmoid(x_seq[:,1:,:]) @ W^T.
// 2000 blocks x 512 threads; each block does 128 (trial,step) rows.
// =====================================================================
#define ROWS_PB   128
#define WOP2_OFF  0                    // 32x256 f16, B-operand order (16 KB)
#define AOP2_OFF  8192                 // 128x256 sigma(x) f16, A-operand order (64 KB)
#define LDS2_HALFS (8192 + 32768)      // 40960 halfs = 81920 bytes

__global__ __launch_bounds__(NTHREADS)
void rnn_readout_kernel(const float* __restrict__ xseq, // (256,1001,256)
                        const float* __restrict__ Wg,   // (32,256)
                        float* __restrict__ out_y)      // (256,1000,32)
{
    extern __shared__ _Float16 smem[];
    _Float16* Wop = smem + WOP2_OFF;
    _Float16* Aop = smem + AOP2_OFF;   // 8 M-tiles x 4096 halfs

    const int tid  = threadIdx.x;
    const int lane = tid & 31;
    const int w    = tid >> 5;
    const int rbase = blockIdx.x * ROWS_PB;   // flattened row = b*1000 + t

    // stage W -> B-operand order
    for (int idx = tid; idx < OUT_DIM * SIZE; idx += NTHREADS) {
        int n = idx >> 8, k = idx & 255;
        Wop[((n >> 4) << 12) + a_addr(n & 15, k)] = (_Float16)Wg[idx];
    }
    // stage sigma(x) for 128 rows -> A-operand order (coalesced k per lane)
    for (int idx = tid; idx < ROWS_PB * SIZE; idx += NTHREADS) {
        int mrow = idx >> 8, k = idx & 255;
        int r = rbase + mrow;
        int b = r / T_STEPS, t = r - b * T_STEPS;
        float xv = xseq[((size_t)b * (T_STEPS + 1) + (t + 1)) * SIZE + k];
        Aop[((mrow >> 4) << 12) + a_addr(mrow & 15, k)] = (_Float16)sigmoidf(xv);
    }
    __syncthreads();

    const int mt = w >> 1;       // M-tile 0..7
    const int nt = w & 1;        // N-tile 0..1
    const _Float16* Ab = Aop + (mt << 12);
    const _Float16* Wb = Wop + (nt << 12);

    v8f c0 = {}, c1 = {};
#pragma unroll
    for (int kk = 0; kk < 8; kk += 2) {
        v16h a0 = *(const v16h*)(Ab + (kk + 0) * 512 + lane * 16);
        v16h b0 = *(const v16h*)(Wb + (kk + 0) * 512 + lane * 16);
        c0 = __builtin_amdgcn_wmma_f32_16x16x32_f16(false, a0, false, b0,
                                                    (short)0, c0, false, false);
        v16h a1 = *(const v16h*)(Ab + (kk + 1) * 512 + lane * 16);
        v16h b1 = *(const v16h*)(Wb + (kk + 1) * 512 + lane * 16);
        c1 = __builtin_amdgcn_wmma_f32_16x16x32_f16(false, a1, false, b1,
                                                    (short)0, c1, false, false);
    }
    v8f c = c0 + c1;

    const int mlo = (lane < 16) ? 0 : 8;
    const int col = nt * 16 + (lane & 15);
#pragma unroll
    for (int v = 0; v < 8; ++v) {
        int r = rbase + mt * 16 + v + mlo;
        out_y[(size_t)r * OUT_DIM + col] = c[v];
    }
}

extern "C" void kernel_launch(void* const* d_in, const int* in_sizes, int n_in,
                              void* d_out, int out_size, void* d_ws, size_t ws_size,
                              hipStream_t stream) {
    const float* inp = (const float*)d_in[0];   // (256,1000,64)
    const float* B   = (const float*)d_in[1];   // (256,64)
    const float* J   = (const float*)d_in[2];   // (256,256)
    const float* W   = (const float*)d_in[3];   // (32,256)
    float* out   = (float*)d_out;
    float* out_x = out;                                            // x_seq
    float* out_y = out + (size_t)N_TRIALS * (T_STEPS + 1) * SIZE;  // y

    // Pass 1: sequential recurrence (writes x_seq)
    hipLaunchKernelGGL(rnn_recur_kernel,
                       dim3(N_TRIALS / TB), dim3(NTHREADS),
                       (size_t)LDS1_HALFS * sizeof(_Float16), stream,
                       inp, B, J, out_x);

    // Pass 2: embarrassingly parallel readout (reads x_seq, writes y)
    hipLaunchKernelGGL(rnn_readout_kernel,
                       dim3((N_TRIALS * T_STEPS) / ROWS_PB), dim3(NTHREADS),
                       (size_t)LDS2_HALFS * sizeof(_Float16), stream,
                       out_x, W, out_y);
}
